// RNN_73177652789314
// MI455X (gfx1250) — compile-verified, matrix operand order checked by reading, not run on hardware
//
#include <hip/hip_runtime.h>
#include <hip/hip_bf16.h>

// ---------------------------------------------------------------------------
// Types for CDNA5 WMMA (wave32): V_WMMA_F32_16X16X32_BF16
// ---------------------------------------------------------------------------
typedef __attribute__((ext_vector_type(16))) __bf16 v16bf;
typedef __attribute__((ext_vector_type(8)))  __bf16 v8bf;
typedef __attribute__((ext_vector_type(4)))  __bf16 v4bf;
typedef __attribute__((ext_vector_type(8)))  float  v8f;
typedef __attribute__((ext_vector_type(4)))  float  v4f;

// ---------------------------------------------------------------------------
// Elementwise helpers
// ---------------------------------------------------------------------------
__global__ void f32_to_bf16_kernel(const float* __restrict__ s,
                                   __bf16* __restrict__ d, int n) {
    int i = blockIdx.x * 256 + threadIdx.x;
    if (i < n) d[i] = (__bf16)s[i];
}

__global__ void zero_bf16_kernel(__bf16* __restrict__ d, int n) {
    int i = blockIdx.x * 256 + threadIdx.x;
    if (i < n) d[i] = (__bf16)0.0f;
}

__global__ void zero_u32_kernel(unsigned int* __restrict__ d, int n) {
    int i = blockIdx.x * 256 + threadIdx.x;
    if (i < n) d[i] = 0u;
}

// ---------------------------------------------------------------------------
// Throughput NT GEMM on WMMA: C[M,N] = A[M,K] * B[N,K]^T + bias.
// WMW x WNW waves per block; wave tile (MW*16) x 64; block tile
// (WMW*MW*16) x (WNW*64). Double-buffered LDS, one barrier per K-step,
// register prefetch of chunk k+1 issued before the WMMA phase of chunk k.
// ---------------------------------------------------------------------------
template<int WMW, int WNW, int MW, bool A_BF16>
__global__ __launch_bounds__(WMW * WNW * 32)
void gemm_nt_wmma(const void* __restrict__ Aptr,
                  const __bf16* __restrict__ B,
                  const float* __restrict__ bias,
                  float* __restrict__ Cout,
                  int Ntot, int Ktot) {
    constexpr int THREADS = WMW * WNW * 32;
    constexpr int BM  = WMW * MW * 16;
    constexpr int BN  = WNW * 64;
    constexpr int ACH = (BM * 32) / (THREADS * 4);
    constexpr int BCH = (BN * 32) / (THREADS * 8);

    // Stride-40 bf16 rows (80B): 16B-aligned b128 slots, conflict-free.
    __shared__ __bf16 As[2][BM][40];
    __shared__ __bf16 Bs[2][BN][40];

    const int tid   = threadIdx.x;
    const int lane  = tid & 31;
    const int wave  = tid >> 5;
    const int wm    = (wave / WNW) * (MW * 16);
    const int wnb   = (wave % WNW) * 64;
    const int bm    = blockIdx.x * BM;
    const int bn    = blockIdx.y * BN;

    const int fr    = lane & 15;
    const int half8 = (lane >> 4) << 3;

    v8f acc[MW][4] = {};

    v4f  apf_f[ACH];
    v4bf apf_b[ACH];
    v8bf bpf[BCH];

    auto prefetch = [&](int k0) {
        #pragma unroll
        for (int i = 0; i < ACH; ++i) {
            const int e = i * THREADS + tid;
            const int r = e >> 3, c = (e & 7) << 2;
            if (A_BF16)
                apf_b[i] = *(const v4bf*)((const __bf16*)Aptr +
                               (size_t)(bm + r) * Ktot + (k0 + c));
            else
                apf_f[i] = *(const v4f*)((const float*)Aptr +
                               (size_t)(bm + r) * Ktot + (k0 + c));
        }
        #pragma unroll
        for (int i = 0; i < BCH; ++i) {
            const int e = i * THREADS + tid;
            const int r = e >> 2, c = (e & 3) << 3;
            bpf[i] = *(const v8bf*)(B + (size_t)(bn + r) * Ktot + (k0 + c));
        }
    };

    auto stage = [&](int buf) {
        #pragma unroll
        for (int i = 0; i < ACH; ++i) {
            const int e = i * THREADS + tid;
            const int r = e >> 3, c = (e & 7) << 2;
            if (A_BF16) {
                *(v4bf*)&As[buf][r][c] = apf_b[i];
            } else {
                v4bf ab;
                ab[0] = (__bf16)apf_f[i][0]; ab[1] = (__bf16)apf_f[i][1];
                ab[2] = (__bf16)apf_f[i][2]; ab[3] = (__bf16)apf_f[i][3];
                *(v4bf*)&As[buf][r][c] = ab;
            }
        }
        #pragma unroll
        for (int i = 0; i < BCH; ++i) {
            const int e = i * THREADS + tid;
            const int r = e >> 2, c = (e & 3) << 3;
            *(v8bf*)&Bs[buf][r][c] = bpf[i];
        }
    };

    prefetch(0);
    stage(0);
    __syncthreads();

    for (int k0 = 0; k0 < Ktot; k0 += 32) {
        const int  cur  = (k0 >> 5) & 1;
        const bool more = (k0 + 32) < Ktot;

        if (more) prefetch(k0 + 32);

        union Frag { v16bf v; v8bf h[2]; };
        Frag fa[MW], fb;
        #pragma unroll
        for (int m = 0; m < MW; ++m) {
            fa[m].h[0] = *(const v8bf*)&As[cur][wm + m * 16 + fr][half8];
            fa[m].h[1] = *(const v8bf*)&As[cur][wm + m * 16 + fr][half8 + 16];
        }
        #pragma unroll
        for (int j = 0; j < 4; ++j) {
            fb.h[0] = *(const v8bf*)&Bs[cur][wnb + j * 16 + fr][half8];
            fb.h[1] = *(const v8bf*)&Bs[cur][wnb + j * 16 + fr][half8 + 16];
            #pragma unroll
            for (int m = 0; m < MW; ++m)
                acc[m][j] = __builtin_amdgcn_wmma_f32_16x16x32_bf16(
                                false, fa[m].v, false, fb.v, (short)0,
                                acc[m][j], false, false);
        }

        if (more) stage(1 - cur);
        __syncthreads();
    }

    const int cmb = bm + wm + ((lane >> 4) << 3);
    #pragma unroll
    for (int m = 0; m < MW; ++m) {
        #pragma unroll
        for (int j = 0; j < 4; ++j) {
            const int cn = bn + wnb + j * 16 + (lane & 15);
            const float bv = bias[cn];
            #pragma unroll
            for (int r = 0; r < 8; ++r) {
                const size_t off = (size_t)(cmb + m * 16 + r) * Ntot + cn;
                Cout[off] = acc[m][j][r] + bv;
            }
        }
    }
}

// ---------------------------------------------------------------------------
// Persistent recurrence kernel: loops all T steps internally; steps are
// separated by a device-scope arrival counter instead of kernel launches.
// Grid: 16 blocks x 128 threads (4 waves). Block b owns N-columns
// [b*64, b*64+64) of h for every step. Per step:
//   h_t = relu(xh_t + h_{t-1} @ Wh^T + bh)
// xh_t lives in `allh` (written by the xh GEMM) and is overwritten with h_t
// (fp32); h_t is also written bf16 into the ping-pong buffer for step t+1.
// All 4 B fragments are preloaded per K-step so the 4 WMMAs issue
// back-to-back (only 4 waves/WGP here -> ds latency must be hidden in-wave).
// ---------------------------------------------------------------------------
__global__ __launch_bounds__(128)
void rnn_persistent(const __bf16* __restrict__ Wh,     // (D, D) bf16
                    const float*  __restrict__ bh,     // (D)
                    float*        __restrict__ allh,   // (T*Nb, D) xh -> h
                    __bf16*       __restrict__ hbuf0,  // (Nb, D) ping (zeroed)
                    __bf16*       __restrict__ hbuf1,  // (Nb, D) pong
                    unsigned int* __restrict__ syncc,  // zeroed counter
                    int Tsteps) {
    constexpr int D  = 1024;
    constexpr int Nb = 64;
    constexpr int NBLK = 16;
    constexpr int THREADS = 128;

    __shared__ __bf16 As[2][64][40];   // h chunk   (64 rows x 32 k)
    __shared__ __bf16 Bs[2][64][40];   // Wh chunk  (64 n    x 32 k)

    const int tid   = threadIdx.x;
    const int lane  = tid & 31;
    const int wave  = tid >> 5;            // 0..3
    const int wm    = wave << 4;           // 16-row A sub-tile
    const int bn    = blockIdx.x << 6;     // 64-column slice of D

    const int fr    = lane & 15;
    const int half8 = (lane >> 4) << 3;

    // Staging regs (128 threads, 64x32 bf16 tiles).
    v4bf apf[4];
    v8bf bpf[2];

    for (int t = 0; t < Tsteps; ++t) {
        const __bf16* hc = (t & 1) ? hbuf1 : hbuf0;
        __bf16*       hn = (t & 1) ? hbuf0 : hbuf1;
        float*        xh = allh + (size_t)t * Nb * D;

        auto prefetch = [&](int k0) {
            #pragma unroll
            for (int i = 0; i < 4; ++i) {
                const int e = i * THREADS + tid;
                const int r = e >> 3, c = (e & 7) << 2;
                apf[i] = *(const v4bf*)(hc + (size_t)r * D + (k0 + c));
            }
            #pragma unroll
            for (int i = 0; i < 2; ++i) {
                const int e = i * THREADS + tid;
                const int r = e >> 2, c = (e & 3) << 3;
                bpf[i] = *(const v8bf*)(Wh + (size_t)(bn + r) * D + (k0 + c));
            }
        };
        auto stage = [&](int buf) {
            #pragma unroll
            for (int i = 0; i < 4; ++i) {
                const int e = i * THREADS + tid;
                const int r = e >> 3, c = (e & 7) << 2;
                *(v4bf*)&As[buf][r][c] = apf[i];
            }
            #pragma unroll
            for (int i = 0; i < 2; ++i) {
                const int e = i * THREADS + tid;
                const int r = e >> 2, c = (e & 3) << 3;
                *(v8bf*)&Bs[buf][r][c] = bpf[i];
            }
        };

        v8f acc[4] = {};

        prefetch(0);
        stage(0);
        __syncthreads();

        for (int k0 = 0; k0 < D; k0 += 32) {
            const int  cur  = (k0 >> 5) & 1;
            const bool more = (k0 + 32) < D;

            if (more) prefetch(k0 + 32);

            // Preload A + all 4 B fragments (10 ds_load_b128 in one burst),
            // then issue the 4 WMMAs back-to-back.
            union Frag { v16bf v; v8bf h[2]; };
            Frag fa, fb[4];
            fa.h[0] = *(const v8bf*)&As[cur][wm + fr][half8];
            fa.h[1] = *(const v8bf*)&As[cur][wm + fr][half8 + 16];
            #pragma unroll
            for (int j = 0; j < 4; ++j) {
                fb[j].h[0] = *(const v8bf*)&Bs[cur][j * 16 + fr][half8];
                fb[j].h[1] = *(const v8bf*)&Bs[cur][j * 16 + fr][half8 + 16];
            }
            #pragma unroll
            for (int j = 0; j < 4; ++j) {
                acc[j] = __builtin_amdgcn_wmma_f32_16x16x32_bf16(
                             false, fa.v, false, fb[j].v, (short)0, acc[j],
                             false, false);
            }

            if (more) stage(1 - cur);
            __syncthreads();
        }

        // Fused epilogue: + xh_t, + bias, relu; fp32 all_h + bf16 h_next.
        const int cmb = wm + ((lane >> 4) << 3);
        #pragma unroll
        for (int j = 0; j < 4; ++j) {
            const int cn = bn + j * 16 + (lane & 15);
            const float bv = bh[cn];
            #pragma unroll
            for (int r = 0; r < 8; ++r) {
                const size_t off = (size_t)(cmb + r) * D + cn;
                float v = acc[j][r] + bv + xh[off];
                v = v > 0.0f ? v : 0.0f;
                xh[off] = v;                 // all_h (fp32), in place
                hn[off] = (__bf16)v;         // next-step operand (bf16)
            }
        }

        // ---- device-scope step barrier ----
        __threadfence();                      // make stores agent-visible
        __syncthreads();                      // whole block done
        if (tid == 0) {
            __hip_atomic_fetch_add(syncc, 1u, __ATOMIC_RELEASE,
                                   __HIP_MEMORY_SCOPE_AGENT);
            const unsigned int want = (unsigned int)NBLK * (t + 1);
            while (__hip_atomic_load(syncc, __ATOMIC_ACQUIRE,
                                     __HIP_MEMORY_SCOPE_AGENT) < want) {
                __builtin_amdgcn_s_sleep(1);
            }
        }
        __syncthreads();                      // release whole block into t+1
    }
}

// ---------------------------------------------------------------------------
// Host-side orchestration (graph-capturable: everything on `stream`).
// Inputs: x (T,N,C) f32 | Wx (D,C) | bx (D) | Wh (D,D) | bh (D) | Wy (K,D) | by (K)
// Output: all_y (T,N,K) f32 then all_h (T,N,D) f32, concatenated.
// ---------------------------------------------------------------------------
extern "C" void kernel_launch(void* const* d_in, const int* in_sizes, int n_in,
                              void* d_out, int out_size, void* d_ws, size_t ws_size,
                              hipStream_t stream) {
    constexpr int T = 512, N = 64, C = 512, D = 1024, K = 512;

    const float* x  = (const float*)d_in[0];
    const float* Wx = (const float*)d_in[1];
    const float* bx = (const float*)d_in[2];
    const float* Wh = (const float*)d_in[3];
    const float* bh = (const float*)d_in[4];
    const float* Wy = (const float*)d_in[5];
    const float* by = (const float*)d_in[6];

    float* y_out = (float*)d_out;                       // (T*N, K)
    float* h_out = y_out + (size_t)T * N * K;           // (T*N, D): xh -> all_h

    // Workspace layout: sync counter | bf16 weights | h ping-pong (~4.5 MB).
    unsigned int* syncc = (unsigned int*)d_ws;
    __bf16* wxb = (__bf16*)((char*)d_ws + 256);         // D*C
    __bf16* whb = wxb + (size_t)D * C;                  // D*D
    __bf16* wyb = whb + (size_t)D * D;                  // K*D
    __bf16* h0  = wyb + (size_t)K * D;                  // N*D
    __bf16* h1  = h0  + (size_t)N * D;                  // N*D

    // 1) Convert weights to bf16; zero h0 and the step counter.
    f32_to_bf16_kernel<<<(D * C + 255) / 256, 256, 0, stream>>>(Wx, wxb, D * C);
    f32_to_bf16_kernel<<<(D * D + 255) / 256, 256, 0, stream>>>(Wh, whb, D * D);
    f32_to_bf16_kernel<<<(K * D + 255) / 256, 256, 0, stream>>>(Wy, wyb, K * D);
    zero_bf16_kernel<<<(N * D + 255) / 256, 256, 0, stream>>>(h0, N * D);
    zero_u32_kernel<<<1, 256, 0, stream>>>(syncc, 1);

    // 2) xh = x @ Wx^T + bx -> written into the all_h region of d_out.
    //    <4,4,2>: 512 thr, block 128x256, 8 wmma/wave/K-step. Grid 256x4.
    gemm_nt_wmma<4, 4, 2, false>
        <<<dim3((T * N) / 128, D / 256), 512, 0, stream>>>(
            x, wxb, bx, h_out, D, C);

    // 3) Recurrence: one persistent kernel, 16 blocks x 128 threads; steps
    //    separated by a device-scope arrival counter (no 512 launches).
    rnn_persistent<<<dim3(16), 128, 0, stream>>>(
        whb, bh, h_out, h0, h1, syncc, T);

    // 4) all_y = all_h @ Wy^T + by. M = 32768, N = 512, K = 1024. Grid 256x2.
    gemm_nt_wmma<4, 4, 2, false>
        <<<dim3((T * N) / 128, K / 256), 512, 0, stream>>>(
            h_out, wyb, by, y_out, K, D);
}
